// Quantize_48000554500147
// MI455X (gfx1250) — matrix operand order repaired and checked by reading, not compiled, not run on hardware
//
#include <hip/hip_runtime.h>
#include <hip/hip_bf16.h>

typedef __bf16 bf16;
typedef __attribute__((ext_vector_type(16))) __bf16 v16bf;
typedef __attribute__((ext_vector_type(8)))  __bf16 v8bf;
typedef __attribute__((ext_vector_type(8)))  float  v8f;
typedef __attribute__((ext_vector_type(4)))  float  v4f;

#define N_ROWS 16384
#define DDIM   512
#define KCB    8192
#define KT     128         // codebook entries per K-tile
#define BLK    256         // 8 wave32
#define EPS    1e-10f

// ---------------------------------------------------------------------------
// Prep kernel: codebook fp32 -> bf16 row-major (cbh), bf16 tile-transposed
// (ct: [K/128][512][128]), and per-entry squared norm (c2w). Runs once per
// call; whole thing is ~16 MB and stays L2-resident for the main kernel.
// ---------------------------------------------------------------------------
__global__ __launch_bounds__(256)
void vq_prep_kernel(const float* __restrict__ cb,
                    bf16* __restrict__ cbh,
                    bf16* __restrict__ ct,
                    float* __restrict__ c2w)
{
    __shared__ float red[256];
    const int k = blockIdx.x;                  // codebook entry
    const float* src = cb + (size_t)k * DDIM;
    bf16* dst_rm = cbh + (size_t)k * DDIM;
    bf16* dst_ct = ct + (size_t)(k >> 7) * DDIM * KT + (k & (KT - 1));
    float sq = 0.0f;
    for (int d = threadIdx.x; d < DDIM; d += 256) {
        float v = src[d];
        bf16 b = (bf16)v;
        dst_rm[d] = b;
        dst_ct[(size_t)d * KT] = b;
        sq += v * v;
    }
    red[threadIdx.x] = sq;
    __syncthreads();
    for (int s = 128; s > 0; s >>= 1) {
        if (threadIdx.x < s) red[threadIdx.x] += red[threadIdx.x + s];
        __syncthreads();
    }
    if (threadIdx.x == 0) c2w[k] = red[0];
}

// Build a 16-element bf16 fragment from two aligned chunks (LDS or global).
__device__ __forceinline__ v16bf make_frag(const bf16* p0, const bf16* p1) {
    v8bf lo = *(const v8bf*)p0;
    v8bf hi = *(const v8bf*)p1;
    return __builtin_shufflevector(lo, hi, 0,1,2,3,4,5,6,7,8,9,10,11,12,13,14,15);
}

// ---------------------------------------------------------------------------
// Main fused kernel: one 16-row x-tile per block, flash-style online softmax
// over K in 128-entry tiles. GEMM1 B and GEMM2 B stream from L2-resident
// bf16 codebook (row-major / tile-transposed) — no LDS staging (zero reuse).
// ---------------------------------------------------------------------------
__global__ __launch_bounds__(BLK)
void Quantize_48000554500147_kernel(const float* __restrict__ x,
                                    const bf16* __restrict__ cbh,
                                    const bf16* __restrict__ ct,
                                    const float* __restrict__ c2w,
                                    const float* __restrict__ gu,
                                    const float* __restrict__ tptr,
                                    float* __restrict__ out)
{
    __shared__ __align__(32) bf16  xA[16 * DDIM];    // 16 KB, x tile bf16
    __shared__ __align__(16) float Sraw[16 * KT];    // 8 KB  raw scores (argmax)
    __shared__ __align__(16) float Slog[16 * KT];    // 8 KB  logits
    __shared__ __align__(16) float Gn[16 * KT];      // 8 KB  gumbel noise
    __shared__ __align__(32) bf16  Wbf[16 * KT];     // 4 KB  softmax weights bf16
    __shared__ float pmax[16 * 16];
    __shared__ float psum[16 * 16];
    __shared__ float pbv[16 * 16];
    __shared__ int   pbi[16 * 16];
    __shared__ float rowM[16], rowL[16], alph[16], bestV[16];
    __shared__ int   bestI[16];

    const int tid  = threadIdx.x;
    const int wave = tid >> 5;
    const int lane = tid & 31;
    const int ln16 = lane & 15;
    const int hi   = lane >> 4;
    const int rowBase = blockIdx.x * 16;
    const float invT = 1.0f / tptr[0];

    if (tid < 16) {
        rowM[tid]  = -INFINITY;
        rowL[tid]  = 0.0f;
        bestV[tid] = -INFINITY;
        bestI[tid] = 0;
    }

    // stage x tile [16 x 512] fp32 -> bf16 LDS (once)
    for (int idx = tid; idx < 16 * DDIM / 4; idx += BLK) {
        int off = idx * 4;
        v4f v = *(const v4f*)(x + (size_t)rowBase * DDIM + off);
        xA[off + 0] = (bf16)v.x; xA[off + 1] = (bf16)v.y;
        xA[off + 2] = (bf16)v.z; xA[off + 3] = (bf16)v.w;
    }

    v8f vzero = {0.f,0.f,0.f,0.f,0.f,0.f,0.f,0.f};
    v8f acc[4];
    acc[0] = vzero; acc[1] = vzero; acc[2] = vzero; acc[3] = vzero;

    __syncthreads();

    for (int kb = 0; kb < KCB; kb += KT) {
        // ============ GEMM1 (waves 0-3) + gumbel noise (waves 4-7) ==========
        if (wave < 4) {
            // wave w computes score columns [w*32, w*32+32): two 16x16 tiles
            v8f s0 = vzero, s1 = vzero;
            const bf16* arow = xA + ln16 * DDIM;
            const bf16* b0r = cbh + (size_t)(kb + wave * 32 + ln16) * DDIM;
            const bf16* b1r = b0r + (size_t)16 * DDIM;
            #pragma unroll
            for (int ks = 0; ks < DDIM / 32; ++ks) {
                int koffA = ks * 32 + hi * 16;    // A: contiguous 16 K/lane-half
                int koffB = ks * 32 + hi * 16;    // B: contiguous 16 K/lane-half
                v16bf a   = make_frag(arow + koffA, arow + koffA + 8);  // LDS
                v16bf bb0 = make_frag(b0r + koffB, b0r + koffB + 8);    // L2
                v16bf bb1 = make_frag(b1r + koffB, b1r + koffB + 8);    // L2
                s0 = __builtin_amdgcn_wmma_f32_16x16x32_bf16(false, a, false, bb0,
                                                             (short)0, s0, false, false);
                s1 = __builtin_amdgcn_wmma_f32_16x16x32_bf16(false, a, false, bb1,
                                                             (short)0, s1, false, false);
            }
            const int col0 = wave * 32 + ln16;
            const int col1 = col0 + 16;
            const float c20 = c2w[kb + col0];
            const float c21 = c2w[kb + col1];
            #pragma unroll
            for (int r = 0; r < 8; ++r) {
                int m = r + hi * 8;
                Sraw[m * KT + col0] = 2.0f * s0[r] - c20;   // x^2 const dropped
                Sraw[m * KT + col1] = 2.0f * s1[r] - c21;
            }
        } else {
            // 128 threads cover 16 rows x 128 cols; 16 contiguous cols each
            int t2 = tid - 128;
            int m = t2 >> 3, q = t2 & 7;
            const float* gsrc = gu + (size_t)(rowBase + m) * KCB + kb + q * 16;
            float* gdst = Gn + m * KT + q * 16;
            #pragma unroll
            for (int i = 0; i < 16; i += 4) {
                v4f u = *(const v4f*)(gsrc + i);
                gdst[i + 0] = -__logf(-__logf(u.x + EPS) + EPS);
                gdst[i + 1] = -__logf(-__logf(u.y + EPS) + EPS);
                gdst[i + 2] = -__logf(-__logf(u.z + EPS) + EPS);
                gdst[i + 3] = -__logf(-__logf(u.w + EPS) + EPS);
            }
        }
        __syncthreads();

        // ============ Phase A: logits + partial max + partial argmax ========
        {
            int m = tid >> 4, q = tid & 15;
            float pm = -INFINITY, bv = -INFINITY;
            int bi = 0;
            #pragma unroll
            for (int j = 0; j < 8; ++j) {
                int col = q * 8 + j;
                float raw = Sraw[m * KT + col];
                float l = (raw + Gn[m * KT + col]) * invT;
                Slog[m * KT + col] = l;
                pm = fmaxf(pm, l);
                if (raw > bv) { bv = raw; bi = kb + col; }
            }
            pmax[m * 16 + q] = pm;
            pbv[m * 16 + q] = bv;
            pbi[m * 16 + q] = bi;
        }
        __syncthreads();

        // ============ Phase B: combine max, alpha, argmax (16 threads) ======
        if (tid < 16) {
            float mOld = rowM[tid];
            float mNew = mOld;
            for (int q = 0; q < 16; ++q) mNew = fmaxf(mNew, pmax[tid * 16 + q]);
            rowM[tid] = mNew;
            alph[tid] = __expf(mOld - mNew);
            float bv = bestV[tid]; int bi = bestI[tid];
            for (int q = 0; q < 16; ++q) {
                float v = pbv[tid * 16 + q];
                if (v > bv) { bv = v; bi = pbi[tid * 16 + q]; }
            }
            bestV[tid] = bv; bestI[tid] = bi;
        }
        __syncthreads();

        // ============ Phase C: weights = exp(l - M), partial sums ===========
        {
            int m = tid >> 4, q = tid & 15;
            float mNew = rowM[m];
            float s = 0.0f;
            #pragma unroll
            for (int j = 0; j < 8; ++j) {
                int col = q * 8 + j;
                float w = __expf(Slog[m * KT + col] - mNew);
                s += w;
                Wbf[m * KT + col] = (bf16)w;
            }
            psum[m * 16 + q] = s;
            if (kb + KT < KCB)   // prefetch next gumbel slice (streaming HBM)
                __builtin_prefetch(gu + (size_t)(rowBase + m) * KCB + kb + KT + q * 8, 0, 0);
        }
        __syncthreads();

        // ============ Phase D: denominator update (16 threads) ==============
        if (tid < 16) {
            float s = 0.0f;
            for (int q = 0; q < 16; ++q) s += psum[tid * 16 + q];
            rowL[tid] = rowL[tid] * alph[tid] + s;
        }
        __syncthreads();

        // ============ GEMM2: acc = acc*alpha + W[16x128] @ C[128 x 64-slice] =
        {
            #pragma unroll
            for (int t = 0; t < 4; ++t)
                #pragma unroll
                for (int r = 0; r < 8; ++r)
                    acc[t][r] *= alph[r + hi * 8];

            const bf16* warow = Wbf + ln16 * KT;
            const bf16* ctb = ct + (size_t)(kb >> 7) * DDIM * KT;
            #pragma unroll
            for (int ks = 0; ks < KT / 32; ++ks) {
                int koffA = ks * 32 + hi * 16;
                int koffB = ks * 32 + hi * 16;
                v16bf a = make_frag(warow + koffA, warow + koffA + 8);   // LDS
                #pragma unroll
                for (int t = 0; t < 4; ++t) {
                    const bf16* brow = ctb + (size_t)(wave * 64 + t * 16 + ln16) * KT;
                    v16bf b = make_frag(brow + koffB, brow + koffB + 8); // L2
                    acc[t] = __builtin_amdgcn_wmma_f32_16x16x32_bf16(false, a, false, b,
                                                                     (short)0, acc[t],
                                                                     false, false);
                }
            }
        }
        // next-iteration barriers provide all WAR protection (see analysis)
    }
    __syncthreads();

    // ---- epilogue: emb = acc / L ; ids ----
    {
        const int dbase = wave * 64;
        #pragma unroll
        for (int t = 0; t < 4; ++t) {
            #pragma unroll
            for (int r = 0; r < 8; ++r) {
                int m = r + hi * 8;
                out[(size_t)(rowBase + m) * DDIM + dbase + t * 16 + ln16] =
                    acc[t][r] / rowL[m];
            }
        }
    }
    if (tid < 16) {
        out[(size_t)N_ROWS * DDIM + rowBase + tid] = (float)bestI[tid];
    }
}

extern "C" void kernel_launch(void* const* d_in, const int* in_sizes, int n_in,
                              void* d_out, int out_size, void* d_ws, size_t ws_size,
                              hipStream_t stream) {
    (void)in_sizes; (void)n_in; (void)out_size; (void)ws_size;
    const float* x  = (const float*)d_in[0];
    const float* cb = (const float*)d_in[1];
    const float* gu = (const float*)d_in[2];
    const float* tp = (const float*)d_in[3];
    float* out = (float*)d_out;

    // workspace carve: bf16 codebook (8 MB) + tile-transposed bf16 (8 MB)
    // + squared norms (32 KB)  => ~16.03 MB, L2-resident on MI455X (192 MB L2)
    bf16*  cbh = (bf16*)d_ws;
    bf16*  ct  = cbh + (size_t)KCB * DDIM;
    float* c2w = (float*)(ct + (size_t)KCB * DDIM);

    vq_prep_kernel<<<dim3(KCB), dim3(256), 0, stream>>>(cb, cbh, ct, c2w);
    Quantize_48000554500147_kernel<<<dim3(N_ROWS / 16), dim3(BLK), 0, stream>>>(
        x, cbh, ct, c2w, gu, tp, out);
}
